// rangePredict_74775380623728
// MI455X (gfx1250) — compile-verified, matrix operand order checked by reading, not run on hardware
//
#include <hip/hip_runtime.h>

// ---------------------------------------------------------------------------
// Problem constants (from reference)
// ---------------------------------------------------------------------------
#define Bn 2
#define Cc 32
#define Dd 12
#define Hh 128
#define Ww 256
#define HW (Hh * Ww)
#define Vv (Bn * Dd * Hh * Ww)      // 786432 logical voxels
#define BHW (Bn * Hh * Ww)          // 65536

// Halo-padded geometry (+1 voxel each side in D,H,W) -> branch-free 3x3x3 taps
#define PD (Dd + 2)                 // 14
#define PH (Hh + 2)                 // 130
#define PW (Ww + 2)                 // 258
#define PV (Bn * PD * PH * PW)      // 939120 padded voxels

typedef __attribute__((ext_vector_type(16))) __bf16 v16bf;
typedef __attribute__((ext_vector_type(8)))  __bf16 v8bf;
typedef __attribute__((ext_vector_type(8)))  float  v8f;

// ---------------------------------------------------------------------------
// bf16 helpers (round-to-nearest-even)
// ---------------------------------------------------------------------------
__device__ __forceinline__ __bf16 f2bf(float f) {
    union { float f; unsigned u; } a; a.f = f;
    unsigned r = a.u + 0x7FFFu + ((a.u >> 16) & 1u);
    union { unsigned short s; __bf16 b; } o; o.s = (unsigned short)(r >> 16);
    return o.b;
}
__device__ __forceinline__ float bf2f(__bf16 b) {
    union { unsigned short s; __bf16 b; } i; i.b = b;
    union { float f; unsigned u; } o; o.u = ((unsigned)i.s) << 16;
    return o.f;
}

// Padded voxel offset for logical (b,z,y,x): halo shift of +1 per spatial dim.
__device__ __forceinline__ size_t pvox(int b, int z, int y, int x) {
    return ((size_t)((b * PD + z + 1) * PH + (y + 1))) * PW + (x + 1);
}

// ---------------------------------------------------------------------------
// Kernel 1: warp + pack cost volume into halo-padded channels-last bf16
// [PV][96]: ch0..31 left, ch32..63 warped right, ch64 samples, ch65..95 zero.
// Halo voxels pre-zeroed by hipMemsetAsync.
// ---------------------------------------------------------------------------
__global__ void pack_kernel(const float* __restrict__ lf,
                            const float* __restrict__ rf,
                            const float* __restrict__ smp,
                            __bf16* __restrict__ out) {
    int v = blockIdx.x * blockDim.x + threadIdx.x;
    if (v >= Vv) return;
    int x = v % Ww; int t = v / Ww;
    int y = t % Hh; t /= Hh;
    int z = t % Dd;
    int b = t / Dd;
    float s = smp[v];
    float fx = (float)x - s;
    if (fx < 0.f) fx = 0.f;
    if (fx > (float)(Ww - 1)) fx = (float)(Ww - 1);
    int rx = (int)fx;                         // trunc == floor (fx >= 0)
    __bf16* o = out + pvox(b, z, y, x) * 96;
    const float* lbase = lf + ((size_t)b * Cc) * HW + (size_t)y * Ww + x;
    const float* rbase = rf + ((size_t)b * Cc) * HW + (size_t)y * Ww + rx;
#pragma unroll 4
    for (int c = 0; c < Cc; ++c) {
        o[c]      = f2bf(lbase[(size_t)c * HW]);
        o[32 + c] = f2bf(rbase[(size_t)c * HW]);
    }
    o[64] = f2bf(s);
#pragma unroll
    for (int c = 65; c < 96; ++c) o[c] = f2bf(0.f);
}

// ---------------------------------------------------------------------------
// Kernel 2: pre-swizzle OIDHW fp32 weights into per-lane WMMA B-fragment
// layout dst[((ntile*K32 + chunk)*32 + lane)*16 + e] (bf16).
// k = tap*CinP + ci, tap = (kz*3+ky)*3+kx; koff(e) = e<8 ? half*8+e
//                                                       : 16+half*8+(e-8).
// ---------------------------------------------------------------------------
__global__ void packw_kernel(const float* __restrict__ w, __bf16* __restrict__ dst,
                             int Cout, int Cin, int CinP, int K32, int NT) {
    int i = blockIdx.x * blockDim.x + threadIdx.x;
    int total = NT * K32 * 32 * 16;
    if (i >= total) return;
    int e     = i & 15;
    int lane  = (i >> 4) & 31;
    int chunk = (i >> 9) % K32;
    int ntile = i / (K32 * 512);
    int half = lane >> 4;
    int n = ntile * 16 + (lane & 15);
    int koff = (e < 8) ? (half * 8 + e) : (16 + half * 8 + (e - 8));
    int k = chunk * 32 + koff;
    int tap = k / CinP, ci = k % CinP;
    float val = 0.f;
    if (n < Cout && ci < Cin) val = w[((size_t)n * Cin + ci) * 27 + tap];
    dst[i] = f2bf(val);
}

// ---------------------------------------------------------------------------
// Kernel 3: implicit-GEMM 3x3x3 conv3d + BN affine + LeakyReLU(0.1).
// Macro-tile per wave: (16*MT) M-voxels (x-contiguous; W % (16*MT) == 0)
// by (16*NTW) N-channels. A-fragments are reused across NTW n-tiles and
// B-fragments across MT m-tiles -> 2*(MT+NTW) b128 loads per MT*NTW wmma
// (2:1 for 32x32 tiles vs 4:1 for 16x16). All tap/chunk/tile address deltas
// are compile-time immediates from one base pointer; halo padding keeps the
// loop branch-free with EXEC always full.
// ---------------------------------------------------------------------------
template <int CINP, int NTALL, int MT, int NTW>
__global__ void conv3d_wmma_kernel(const __bf16* __restrict__ in,   // [PV][CINP]
                                   const __bf16* __restrict__ wpk,
                                   const float* __restrict__ gamma,
                                   const float* __restrict__ beta,
                                   __bf16* __restrict__ out,        // [PV][CoutP]
                                   int CoutP) {
    constexpr int CH = CINP / 32;          // K-chunks per tap
    constexpr int NG = NTALL / NTW;        // n-groups per m-group
    const int wave = blockIdx.x * (blockDim.x >> 5) + (threadIdx.x >> 5);
    const int lane = threadIdx.x & 31;
    const int totalWaves = (Vv / (16 * MT)) * NG;
    if (wave >= totalWaves) return;        // grid sized exactly; never trips
    const int mg = wave / NG;
    const int ng = wave % NG;
    const int half = lane >> 4;
    const int mr = lane & 15;
    const int m = mg * (16 * MT) + mr;     // lane's row in first m-tile
    const int x = m % Ww; int t = m / Ww;
    const int y = t % Hh; t /= Hh;
    const int z = t % Dd;
    const int b = t / Dd;

    // Base = padded address of tap (kz,ky,kx)=(0,0,0) == logical (z-1,y-1,x-1)
    const __bf16* base =
        in + ((size_t)((b * PD + z) * PH + y) * PW + x) * CINP + half * 8;
    const __bf16* wp =
        wpk + ((size_t)(ng * NTW) * (27 * CH)) * 512 + lane * 16;

    union Frag { v16bf v; v8bf h[2]; };
    const v8f vzero = {};
    v8f acc[MT][NTW];
#pragma unroll
    for (int i = 0; i < MT; ++i)
#pragma unroll
        for (int j = 0; j < NTW; ++j) acc[i][j] = vzero;

#pragma unroll
    for (int kz = 0; kz < 3; ++kz) {
#pragma unroll
        for (int ky = 0; ky < 3; ++ky) {
#pragma unroll
            for (int kx = 0; kx < 3; ++kx) {
                const int tap = (kz * 3 + ky) * 3 + kx;
                const __bf16* p =
                    base + (size_t)((kz * PH + ky) * PW + kx) * CINP;
#pragma unroll
                for (int c = 0; c < CH; ++c) {
                    Frag A[MT], Bf[NTW];
#pragma unroll
                    for (int i = 0; i < MT; ++i) {
                        const __bf16* pa = p + i * 16 * CINP + c * 32;
                        A[i].h[0] = *(const v8bf*)pa;          // K run 1
                        A[i].h[1] = *(const v8bf*)(pa + 16);   // K run 2
                    }
#pragma unroll
                    for (int j = 0; j < NTW; ++j) {
                        const __bf16* q =
                            wp + (size_t)(j * 27 * CH + tap * CH + c) * 512;
                        Bf[j].h[0] = *(const v8bf*)q;
                        Bf[j].h[1] = *(const v8bf*)(q + 8);
                    }
#pragma unroll
                    for (int i = 0; i < MT; ++i)
#pragma unroll
                        for (int j = 0; j < NTW; ++j)
                            acc[i][j] = __builtin_amdgcn_wmma_f32_16x16x32_bf16(
                                false, A[i].v, false, Bf[j].v, (short)0,
                                acc[i][j], false, false);
                }
            }
        }
    }

    // Epilogue: D layout -> lane n = lane&15, rows half*8 + r (x-contiguous).
    const int x0 = x - mr;                 // macro-tile's first x
    const size_t orow = ((size_t)((b * PD + z + 1) * PH + (y + 1))) * PW;
#pragma unroll
    for (int j = 0; j < NTW; ++j) {
        const int n = (ng * NTW + j) * 16 + mr;
        const float g  = gamma[n];
        const float bb = beta[n];
#pragma unroll
        for (int i = 0; i < MT; ++i) {
#pragma unroll
            for (int r = 0; r < 8; ++r) {
                float yv = acc[i][j][r] * g + bb;
                yv = yv > 0.f ? yv : 0.1f * yv;    // LeakyReLU(0.1)
                const int xo = x0 + i * 16 + half * 8 + r;
                out[(orow + xo + 1) * CoutP + n] = f2bf(yv);
            }
        }
    }
}

// ---------------------------------------------------------------------------
// Kernel 4: 16->1 3x3x3 conv (score head), branch-free over padded t16.
// ---------------------------------------------------------------------------
__global__ void conv3d_16to1_kernel(const __bf16* __restrict__ tin,  // [PV][16]
                                    const float* __restrict__ w2,
                                    const float* __restrict__ b2,
                                    float* __restrict__ sc) {
    int v = blockIdx.x * blockDim.x + threadIdx.x;
    if (v >= Vv) return;
    int x = v % Ww; int t = v / Ww;
    int y = t % Hh; t /= Hh;
    int z = t % Dd;
    int b = t / Dd;
    const __bf16* base =
        tin + ((size_t)((b * PD + z) * PH + y) * PW + x) * 16;
    float acc = b2[0];
#pragma unroll
    for (int kz = 0; kz < 3; ++kz)
#pragma unroll
        for (int ky = 0; ky < 3; ++ky)
#pragma unroll
            for (int kx = 0; kx < 3; ++kx) {
                const int tap = (kz * 3 + ky) * 3 + kx;
                const __bf16* p = base + (size_t)((kz * PH + ky) * PW + kx) * 16;
#pragma unroll
                for (int ci = 0; ci < 16; ++ci)
                    acc += bf2f(p[ci]) * w2[ci * 27 + tap];
            }
    sc[v] = acc;
}

// ---------------------------------------------------------------------------
// Kernel 5: softmax over D (sign applied) + disparity expectation.
// ---------------------------------------------------------------------------
__global__ void softmax_kernel(const float* __restrict__ sc,
                               const float* __restrict__ smp,
                               float sign,
                               float* __restrict__ prob,
                               float* __restrict__ disp) {
    int p = blockIdx.x * blockDim.x + threadIdx.x;
    if (p >= BHW) return;
    int x = p % Ww; int t = p / Ww;
    int y = t % Hh;
    int b = t / Hh;
    size_t base = ((size_t)b * Dd) * HW + (size_t)y * Ww + x;
    float s[Dd];
    float mx = -3.0e38f;
#pragma unroll
    for (int d = 0; d < Dd; ++d) {
        s[d] = sign * sc[base + (size_t)d * HW];
        mx = s[d] > mx ? s[d] : mx;
    }
    float sum = 0.f;
#pragma unroll
    for (int d = 0; d < Dd; ++d) { s[d] = __expf(s[d] - mx); sum += s[d]; }
    float inv = 1.f / sum;
    float dd = 0.f;
#pragma unroll
    for (int d = 0; d < Dd; ++d) {
        float pr = s[d] * inv;
        prob[base + (size_t)d * HW] = pr;
        dd += pr * smp[base + (size_t)d * HW];
    }
    disp[p] = dd;
}

// ---------------------------------------------------------------------------
// Kernel 6: 5x5 2D conv, 1->1 channel, pad 2, + bias + ReLU.
// ---------------------------------------------------------------------------
__global__ void disp_conv_kernel(const float* __restrict__ din,
                                 const float* __restrict__ w,
                                 const float* __restrict__ bias,
                                 float* __restrict__ dout) {
    int p = blockIdx.x * blockDim.x + threadIdx.x;
    if (p >= BHW) return;
    int x = p % Ww; int t = p / Ww;
    int y = t % Hh;
    int b = t / Hh;
    float acc = bias[0];
#pragma unroll
    for (int ky = 0; ky < 5; ++ky) {
        int iy = y + ky - 2;
        if ((unsigned)iy >= (unsigned)Hh) continue;
#pragma unroll
        for (int kx = 0; kx < 5; ++kx) {
            int ix = x + kx - 2;
            if ((unsigned)ix >= (unsigned)Ww) continue;
            acc += din[((size_t)b * Hh + iy) * Ww + ix] * w[ky * 5 + kx];
        }
    }
    dout[p] = acc > 0.f ? acc : 0.f;
}

// ---------------------------------------------------------------------------
// Kernel 7: 5x5 2D conv 12->12, pad 2, + bias, BN affine, LeakyReLU.
// ---------------------------------------------------------------------------
__global__ void feat_conv_kernel(const float* __restrict__ prob,
                                 const float* __restrict__ w,
                                 const float* __restrict__ bias,
                                 const float* __restrict__ g,
                                 const float* __restrict__ be,
                                 float* __restrict__ outp) {
    int i = blockIdx.x * blockDim.x + threadIdx.x;
    if (i >= Vv) return;
    int x = i % Ww; int t = i / Ww;
    int y = t % Hh; t /= Hh;
    int co = t % Dd;
    int b  = t / Dd;
    float acc = bias[co];
    for (int ci = 0; ci < Dd; ++ci) {
        const float* pin = prob + ((size_t)b * Dd + ci) * HW;
        const float* wk  = w + ((size_t)co * Dd + ci) * 25;
#pragma unroll
        for (int ky = 0; ky < 5; ++ky) {
            int iy = y + ky - 2;
            if ((unsigned)iy >= (unsigned)Hh) continue;
#pragma unroll
            for (int kx = 0; kx < 5; ++kx) {
                int ix = x + kx - 2;
                if ((unsigned)ix >= (unsigned)Ww) continue;
                acc += pin[(size_t)iy * Ww + ix] * wk[ky * 5 + kx];
            }
        }
    }
    acc = acc * g[co] + be[co];
    acc = acc > 0.f ? acc : 0.1f * acc;
    outp[i] = acc;
}

// ---------------------------------------------------------------------------
// Host orchestration.
// Input flattening assumption: [left, right, samples] then params pytree
// (alphabetical at each dict level; leaf dicts give b,g,w / b,w / b,be,g,w).
// ---------------------------------------------------------------------------
enum {
    IN_LEFT = 0, IN_RIGHT = 1, IN_SAMPLES = 2,
    IN_D0A_B = 3, IN_D0A_G = 4, IN_D0A_W = 5,
    IN_D0B_B = 6, IN_D0B_G = 7, IN_D0B_W = 8,
    IN_D1A_B = 9, IN_D1A_G = 10, IN_D1A_W = 11,
    IN_D1B_B = 12, IN_D1B_G = 13, IN_D1B_W = 14,
    IN_MAXC1_B = 15, IN_MAXC1_G = 16, IN_MAXC1_W = 17,
    IN_MAXC2B = 18, IN_MAXC2W = 19,
    IN_MAXDISP_B = 20, IN_MAXDISP_W = 21,
    IN_MAXFEAT_B = 22, IN_MAXFEAT_BE = 23, IN_MAXFEAT_G = 24, IN_MAXFEAT_W = 25,
    IN_MINC1_B = 26, IN_MINC1_G = 27, IN_MINC1_W = 28,
    IN_MINC2B = 29, IN_MINC2W = 30,
    IN_MINDISP_B = 31, IN_MINDISP_W = 32,
    IN_MINFEAT_B = 33, IN_MINFEAT_BE = 34, IN_MINFEAT_G = 35, IN_MINFEAT_W = 36,
};

extern "C" void kernel_launch(void* const* d_in, const int* in_sizes, int n_in,
                              void* d_out, int out_size, void* d_ws, size_t ws_size,
                              hipStream_t stream) {
    (void)in_sizes; (void)n_in; (void)out_size; (void)ws_size;

    const float* lf  = (const float*)d_in[IN_LEFT];
    const float* rf  = (const float*)d_in[IN_RIGHT];
    const float* smp = (const float*)d_in[IN_SAMPLES];

    char* ws = (char*)d_ws;
    size_t off = 0;
    auto take = [&](size_t bytes) {
        size_t r = off;
        off += (bytes + 255) & ~(size_t)255;
        return r;
    };

    // Two ping-pong activation regions (halo-padded, channels-last bf16).
    const size_t SZ_A = (size_t)PV * 96 * 2;   // packed(96) / a32(32) / a16p(32)
    const size_t SZ_B = (size_t)PV * 64 * 2;   // a64(64) / a32b(32) / t16(16)
    size_t regionA = take(SZ_A);
    size_t regionB = take(SZ_B);

    size_t wpk_d0a = take((size_t)4 * 81 * 512 * 2);
    size_t wpk_d0b = take((size_t)2 * 54 * 512 * 2);
    size_t wpk_d1a = take((size_t)2 * 27 * 512 * 2);
    size_t wpk_d1b = take((size_t)1 * 27 * 512 * 2);
    size_t wpk_c1n = take((size_t)1 * 27 * 512 * 2);
    size_t wpk_c1x = take((size_t)1 * 27 * 512 * 2);

    size_t sc_min = take((size_t)Vv * 4);
    size_t sc_max = take((size_t)Vv * 4);
    size_t pr_min = take((size_t)Vv * 4);
    size_t pr_max = take((size_t)Vv * 4);
    size_t dp_min = take((size_t)BHW * 4);
    size_t dp_max = take((size_t)BHW * 4);

    __bf16* packed = (__bf16*)(ws + regionA);
    __bf16* a64    = (__bf16*)(ws + regionB);
    __bf16* a32    = (__bf16*)(ws + regionA);   // reuses packed (dead)
    __bf16* a32b   = (__bf16*)(ws + regionB);   // reuses a64 (dead)
    __bf16* a16p   = (__bf16*)(ws + regionA);   // reuses a32 (dead)
    __bf16* t16    = (__bf16*)(ws + regionB);   // reuses a32b (dead)

    const int TPB = 256;
    const int WPB = TPB / 32;                   // wave32: 8 waves per block
    auto wblocks = [&](long waves) { return (int)((waves + WPB - 1) / WPB); };

    // 1) Pack cost volume (halo zero-fill, then interior writes).
    hipMemsetAsync(packed, 0, (size_t)PV * 96 * 2, stream);
    pack_kernel<<<Vv / TPB, TPB, 0, stream>>>(lf, rf, smp, packed);

    // 2) Weight swizzles (tiny).
    packw_kernel<<<(4 * 81 * 512) / TPB, TPB, 0, stream>>>(
        (const float*)d_in[IN_D0A_W], (__bf16*)(ws + wpk_d0a), 64, 65, 96, 81, 4);
    packw_kernel<<<(2 * 54 * 512) / TPB, TPB, 0, stream>>>(
        (const float*)d_in[IN_D0B_W], (__bf16*)(ws + wpk_d0b), 32, 64, 64, 54, 2);
    packw_kernel<<<(2 * 27 * 512) / TPB, TPB, 0, stream>>>(
        (const float*)d_in[IN_D1A_W], (__bf16*)(ws + wpk_d1a), 32, 32, 32, 27, 2);
    packw_kernel<<<(1 * 27 * 512) / TPB, TPB, 0, stream>>>(
        (const float*)d_in[IN_D1B_W], (__bf16*)(ws + wpk_d1b), 16, 32, 32, 27, 1);
    packw_kernel<<<(1 * 27 * 512) / TPB, TPB, 0, stream>>>(
        (const float*)d_in[IN_MINC1_W], (__bf16*)(ws + wpk_c1n), 16, 16, 32, 27, 1);
    packw_kernel<<<(1 * 27 * 512) / TPB, TPB, 0, stream>>>(
        (const float*)d_in[IN_MAXC1_W], (__bf16*)(ws + wpk_c1x), 16, 16, 32, 27, 1);

    // 3) dres0 / dres1 WMMA conv stacks (memset clears halos of each output).
    // 32x32 macro-tiles where Cout >= 32, 32x16 for Cout == 16.
    hipMemsetAsync(a64, 0, (size_t)PV * 64 * 2, stream);
    conv3d_wmma_kernel<96, 4, 2, 2>
        <<<wblocks((long)(Vv / 32) * 2), TPB, 0, stream>>>(
        packed, (const __bf16*)(ws + wpk_d0a),
        (const float*)d_in[IN_D0A_G], (const float*)d_in[IN_D0A_B], a64, 64);
    hipMemsetAsync(a32, 0, (size_t)PV * 32 * 2, stream);
    conv3d_wmma_kernel<64, 2, 2, 2>
        <<<wblocks((long)(Vv / 32)), TPB, 0, stream>>>(
        a64, (const __bf16*)(ws + wpk_d0b),
        (const float*)d_in[IN_D0B_G], (const float*)d_in[IN_D0B_B], a32, 32);
    hipMemsetAsync(a32b, 0, (size_t)PV * 32 * 2, stream);
    conv3d_wmma_kernel<32, 2, 2, 2>
        <<<wblocks((long)(Vv / 32)), TPB, 0, stream>>>(
        a32, (const __bf16*)(ws + wpk_d1a),
        (const float*)d_in[IN_D1A_G], (const float*)d_in[IN_D1A_B], a32b, 32);
    // d1b writes 16 real of 32 padded channels -> memset covers pad + halo.
    hipMemsetAsync(a16p, 0, (size_t)PV * 32 * 2, stream);
    conv3d_wmma_kernel<32, 1, 2, 1>
        <<<wblocks((long)(Vv / 32)), TPB, 0, stream>>>(
        a32b, (const __bf16*)(ws + wpk_d1b),
        (const float*)d_in[IN_D1B_G], (const float*)d_in[IN_D1B_B], a16p, 32);

    float* out_mindisp = (float*)d_out;
    float* out_maxdisp = (float*)d_out + BHW;
    float* out_minfeat = (float*)d_out + 2 * BHW;
    float* out_maxfeat = (float*)d_out + 2 * BHW + Vv;

    // 4) MIN head: c1 (WMMA) -> c2 (scalar) -> softmax(-scores).
    hipMemsetAsync(t16, 0, (size_t)PV * 16 * 2, stream);
    conv3d_wmma_kernel<32, 1, 2, 1>
        <<<wblocks((long)(Vv / 32)), TPB, 0, stream>>>(
        a16p, (const __bf16*)(ws + wpk_c1n),
        (const float*)d_in[IN_MINC1_G], (const float*)d_in[IN_MINC1_B], t16, 16);
    conv3d_16to1_kernel<<<Vv / TPB, TPB, 0, stream>>>(
        t16, (const float*)d_in[IN_MINC2W], (const float*)d_in[IN_MINC2B],
        (float*)(ws + sc_min));
    softmax_kernel<<<BHW / TPB, TPB, 0, stream>>>(
        (const float*)(ws + sc_min), smp, -1.0f,
        (float*)(ws + pr_min), (float*)(ws + dp_min));

    // 5) MAX head (t16 reuse; interior fully overwritten, halos stay zero).
    conv3d_wmma_kernel<32, 1, 2, 1>
        <<<wblocks((long)(Vv / 32)), TPB, 0, stream>>>(
        a16p, (const __bf16*)(ws + wpk_c1x),
        (const float*)d_in[IN_MAXC1_G], (const float*)d_in[IN_MAXC1_B], t16, 16);
    conv3d_16to1_kernel<<<Vv / TPB, TPB, 0, stream>>>(
        t16, (const float*)d_in[IN_MAXC2W], (const float*)d_in[IN_MAXC2B],
        (float*)(ws + sc_max));
    softmax_kernel<<<BHW / TPB, TPB, 0, stream>>>(
        (const float*)(ws + sc_max), smp, 1.0f,
        (float*)(ws + pr_max), (float*)(ws + dp_max));

    // 6) Disparity refinement 5x5 convs (+ReLU) -> outputs 0,1.
    disp_conv_kernel<<<BHW / TPB, TPB, 0, stream>>>(
        (const float*)(ws + dp_min), (const float*)d_in[IN_MINDISP_W],
        (const float*)d_in[IN_MINDISP_B], out_mindisp);
    disp_conv_kernel<<<BHW / TPB, TPB, 0, stream>>>(
        (const float*)(ws + dp_max), (const float*)d_in[IN_MAXDISP_W],
        (const float*)d_in[IN_MAXDISP_B], out_maxdisp);

    // 7) Feature 5x5 convs (+bias+BN+LeakyReLU) -> outputs 2,3.
    feat_conv_kernel<<<Vv / TPB, TPB, 0, stream>>>(
        (const float*)(ws + pr_min), (const float*)d_in[IN_MINFEAT_W],
        (const float*)d_in[IN_MINFEAT_B], (const float*)d_in[IN_MINFEAT_G],
        (const float*)d_in[IN_MINFEAT_BE], out_minfeat);
    feat_conv_kernel<<<Vv / TPB, TPB, 0, stream>>>(
        (const float*)(ws + pr_max), (const float*)d_in[IN_MAXFEAT_W],
        (const float*)d_in[IN_MAXFEAT_B], (const float*)d_in[IN_MAXFEAT_G],
        (const float*)d_in[IN_MAXFEAT_BE], out_maxfeat);
}
